// SoftMaxAttn_64450279244020
// MI455X (gfx1250) — matrix operand order repaired and checked by reading, not compiled, not run on hardware
//
#include <hip/hip_runtime.h>

// ---------------- problem constants ----------------
#define B_    2
#define T_    1024
#define C_    6144
#define H_    64
#define HK_   8
#define D_    128
#define NREP_ (H_ / HK_)
#define QD_   (H_ * D_)   // 8192
#define KD_   (HK_ * D_)  // 1024
#define SCALE_ 0.08838834764831845f  // 1/sqrt(128)

typedef __attribute__((ext_vector_type(16))) __bf16 v16bf;
typedef __attribute__((ext_vector_type(8)))  float  v8f;
typedef unsigned __attribute__((ext_vector_type(4))) v4u;
typedef unsigned __attribute__((ext_vector_type(8))) v8u;

union Frag { uint4 u[2]; v16bf v; };

__device__ __forceinline__ unsigned short f2bf(float f) {
  unsigned u = __float_as_uint(f);
  u += 0x7fffu + ((u >> 16) & 1u);   // round-to-nearest-even
  return (unsigned short)(u >> 16);
}

__device__ __forceinline__ v8f wmma_bf16(v16bf a, v16bf b, v8f c) {
  return __builtin_amdgcn_wmma_f32_16x16x32_bf16(false, a, false, b, (short)0, c,
                                                 false, false);
}

__device__ __forceinline__ void wait_tensorcnt0() {
#if __has_builtin(__builtin_amdgcn_s_wait_tensorcnt)
  __builtin_amdgcn_s_wait_tensorcnt(0);
#else
  asm volatile("s_wait_tensorcnt 0x0" ::: "memory");
#endif
}

// --- Tensor Data Mover: 2D tile global->LDS (D# groups 0-1, 2-group form) ----
// data_size = 2 bytes. tile = tile_d0 (elems) x tile_d1 (rows), row stride =
// stride_elems in global memory; LDS destination is fully packed/contiguous.
__device__ __forceinline__ void tdm_load_2d(unsigned lds_off,
                                            const unsigned short* gptr,
                                            unsigned tile_d0, unsigned tile_d1,
                                            unsigned stride_elems) {
  unsigned long long ga = (unsigned long long)gptr;
  v4u g0;
  g0[0] = 1u;                                   // count=1, user mode, no gather
  g0[1] = lds_off;                              // lds_addr (bytes)
  g0[2] = (unsigned)(ga & 0xffffffffu);         // global_addr[31:0]
  g0[3] = (unsigned)((ga >> 32) & 0x01ffffffu)  // global_addr[56:32]
          | (2u << 30);                         // type = 2 ("image")
  v8u g1;
  g1[0] = (1u << 16);                           // data_size=1 (2B), no flags
  g1[1] = (tile_d0 & 0xffffu) << 16;            // tensor_dim0[15:0]
  g1[2] = ((tile_d0 >> 16) & 0xffffu)           // tensor_dim0[31:16]
          | ((tile_d1 & 0xffffu) << 16);        // tensor_dim1[15:0]
  g1[3] = ((tile_d1 >> 16) & 0xffffu)           // tensor_dim1[31:16]
          | ((tile_d0 & 0xffffu) << 16);        // tile_dim0
  g1[4] = (tile_d1 & 0xffffu);                  // tile_dim1 (tile_dim2 = 0)
  g1[5] = stride_elems;                         // tensor_dim0_stride[31:0]
  g1[6] = 0u;                                   // stride[47:32]=0, dim1_stride lo
  g1[7] = 0u;                                   // dim1_stride hi
  asm volatile("tensor_load_to_lds %0, %1" :: "s"(g0), "s"(g1) : "memory");
}

// ---------------- fp32 -> bf16 convert ----------------
struct alignas(8) US4 { unsigned short s[4]; };

__global__ __launch_bounds__(256) void cvt_f32_bf16(const float* __restrict__ src,
                                                    unsigned short* __restrict__ dst,
                                                    long n4) {
  long i = (long)blockIdx.x * blockDim.x + threadIdx.x;
  if (i >= n4) return;
  float4 f = ((const float4*)src)[i];
  US4 o;
  o.s[0] = f2bf(f.x); o.s[1] = f2bf(f.y);
  o.s[2] = f2bf(f.z); o.s[3] = f2bf(f.w);
  ((US4*)dst)[i] = o;
}

// ---------------- GEMM: C[M,N] = A[M,K] * W[N,K]^T (bf16 in, fp32 out) -------
// block = 256 threads (8 waves). Tile 128(M) x 128(N), K-step 64, TDM-staged
// double-buffered LDS. Each wave DMAs its own 16-row slice of the A and W
// tiles and tracks them on its own TENSORcnt.
__global__ __launch_bounds__(256) void gemm_bf16_tn(const unsigned short* __restrict__ A,
                                                    const unsigned short* __restrict__ W,
                                                    float* __restrict__ Cout,
                                                    int M, int N, int K) {
  __shared__ unsigned short As[2][128 * 64];
  __shared__ unsigned short Ws[2][128 * 64];

  const int tid  = threadIdx.x;
  const int lane = tid & 31;
  const int wave = tid >> 5;
  const int wm   = wave >> 2;   // 0..1
  const int wn   = wave & 3;    // 0..3
  const int l15  = lane & 15;
  const int lHi  = lane >> 4;
  const int mBase = blockIdx.y * 128;
  const int nBase = blockIdx.x * 128;

  v8f acc[4][2];
#pragma unroll
  for (int i = 0; i < 4; ++i)
#pragma unroll
    for (int j = 0; j < 2; ++j) acc[i][j] = (v8f)0.0f;

  // per-wave TDM issue of one K-step's tiles into buffer b
  auto issue = [&](int b, int kk) {
    unsigned ldsA = (unsigned)(unsigned long long)(void*)&As[b][wave * 16 * 64];
    unsigned ldsW = (unsigned)(unsigned long long)(void*)&Ws[b][wave * 16 * 64];
    tdm_load_2d(ldsA, A + (size_t)(mBase + wave * 16) * K + kk, 64, 16, (unsigned)K);
    tdm_load_2d(ldsW, W + (size_t)(nBase + wave * 16) * K + kk, 64, 16, (unsigned)K);
  };

  issue(0, 0);
  wait_tensorcnt0();
  __syncthreads();

  int buf = 0;
  for (int kk = 0; kk < K; kk += 64) {
    if (kk + 64 < K) issue(buf ^ 1, kk + 64);

#pragma unroll
    for (int kc = 0; kc < 64; kc += 32) {
      Frag af[4];
#pragma unroll
      for (int i = 0; i < 4; ++i) {
        const unsigned short* base = &As[buf][(wm * 64 + i * 16 + l15) * 64 + kc];
        af[i].u[0] = *(const uint4*)(base + lHi * 8);
        af[i].u[1] = *(const uint4*)(base + lHi * 8 + 16);
      }
      Frag bfr[2];
#pragma unroll
      for (int j = 0; j < 2; ++j) {
        const unsigned short* base = &Ws[buf][(wn * 32 + j * 16 + l15) * 64 + kc];
        bfr[j].u[0] = *(const uint4*)(base + lHi * 8);
        bfr[j].u[1] = *(const uint4*)(base + lHi * 8 + 16);
      }
#pragma unroll
      for (int i = 0; i < 4; ++i)
#pragma unroll
        for (int j = 0; j < 2; ++j)
          acc[i][j] = wmma_bf16(af[i].v, bfr[j].v, acc[i][j]);
    }

    wait_tensorcnt0();   // our next-buffer DMAs done (no-op on last iter)
    __syncthreads();     // everyone's reads done + everyone's slices landed
    buf ^= 1;
  }

#pragma unroll
  for (int i = 0; i < 4; ++i)
#pragma unroll
    for (int j = 0; j < 2; ++j)
#pragma unroll
      for (int g = 0; g < 8; ++g) {
        int row = mBase + wm * 64 + i * 16 + lHi * 8 + g;
        int col = nBase + wn * 32 + j * 16 + l15;
        Cout[(size_t)row * N + col] = acc[i][j][g];
      }
}

// ---------------- RoPE on Q (also folds 1/sqrt(D) scale), fp32 -> bf16 -------
__global__ __launch_bounds__(256) void rope_q_kernel(const float* __restrict__ q,
                                                     const float* __restrict__ cosT,
                                                     const float* __restrict__ sinT,
                                                     unsigned short* __restrict__ qb) {
  long idx = (long)blockIdx.x * blockDim.x + threadIdx.x;  // over B*T*H*(D/2)
  if (idx >= (long)B_ * T_ * H_ * (D_ / 2)) return;
  int  i = (int)(idx & 63);
  long r = idx >> 6;                 // (b*T + t)*H + h
  int  t = (int)((r / H_) % T_);
  float c = cosT[t * 64 + i], s = sinT[t * 64 + i];
  float xr = q[r * D_ + 2 * i], xi = q[r * D_ + 2 * i + 1];
  qb[r * D_ + 2 * i]     = f2bf((xr * c - xi * s) * SCALE_);
  qb[r * D_ + 2 * i + 1] = f2bf((xr * s + xi * c) * SCALE_);
}

// ---------------- RoPE on K, repack [B,T,HK,D] -> [B,HK,T,D] bf16 ------------
__global__ __launch_bounds__(256) void rope_k_kernel(const float* __restrict__ k,
                                                     const float* __restrict__ cosT,
                                                     const float* __restrict__ sinT,
                                                     unsigned short* __restrict__ kb) {
  long idx = (long)blockIdx.x * blockDim.x + threadIdx.x;  // over B*T*HK*(D/2)
  if (idx >= (long)B_ * T_ * HK_ * (D_ / 2)) return;
  int  i  = (int)(idx & 63);
  long r  = idx >> 6;                // (b*T + t)*HK + hk
  int  hk = (int)(r % HK_);
  long bt = r / HK_;
  int  t  = (int)(bt % T_);
  int  b  = (int)(bt / T_);
  float c = cosT[t * 64 + i], s = sinT[t * 64 + i];
  float xr = k[r * D_ + 2 * i], xi = k[r * D_ + 2 * i + 1];
  size_t o = ((size_t)(b * HK_ + hk) * T_ + t) * D_;
  kb[o + 2 * i]     = f2bf(xr * c - xi * s);
  kb[o + 2 * i + 1] = f2bf(xr * s + xi * c);
}

// ---------------- V pack+transpose: [B,T,HK,D] fp32 -> [B,HK,D,T] bf16 -------
__global__ __launch_bounds__(256) void vt_pack_kernel(const float* __restrict__ v,
                                                      unsigned short* __restrict__ vtb) {
  long idx = (long)blockIdx.x * blockDim.x + threadIdx.x;  // over B*T*HK*D
  if (idx >= (long)B_ * T_ * HK_ * D_) return;
  int  d  = (int)(idx & (D_ - 1));
  long r  = idx >> 7;                // (b*T + t)*HK + hk
  int  hk = (int)(r % HK_);
  long bt = r / HK_;
  int  t  = (int)(bt % T_);
  int  b  = (int)(bt / T_);
  vtb[((size_t)(b * HK_ + hk) * D_ + d) * T_ + t] = f2bf(v[idx]);
}

// ---------------- Flash attention (causal, online softmax) -------------------
// grid = (T/64, B*H), block = 128 (4 waves). Wave w owns 16 q-rows.
__global__ __launch_bounds__(128) void flash_attn(const unsigned short* __restrict__ Qb,
                                                  const unsigned short* __restrict__ Kb,
                                                  const unsigned short* __restrict__ Vt,
                                                  unsigned short* __restrict__ Ob) {
  __shared__ unsigned short Pl[4 * 16 * 64];  // per-wave 16x64 P staging

  const int qt   = blockIdx.x;           // q tile (64 rows)
  const int bh   = blockIdx.y;
  const int b    = bh / H_;
  const int h    = bh % H_;
  const int hk   = h / NREP_;
  const int wave = threadIdx.x >> 5;
  const int lane = threadIdx.x & 31;
  const int l15  = lane & 15;
  const int lHi  = lane >> 4;
  const int qRowBase = qt * 64 + wave * 16;

  // Q fragments held in registers for the whole kernel (4 K-steps of 32)
  Frag qa[4];
  {
    int tq = qRowBase + l15;
    const unsigned short* qp = Qb + ((size_t)(b * T_ + tq) * H_ + h) * D_;
#pragma unroll
    for (int ks = 0; ks < 4; ++ks) {
      qa[ks].u[0] = *(const uint4*)(qp + ks * 32 + lHi * 8);
      qa[ks].u[1] = *(const uint4*)(qp + ks * 32 + lHi * 8 + 16);
    }
  }

  v8f o[8];
#pragma unroll
  for (int d8 = 0; d8 < 8; ++d8) o[d8] = (v8f)0.0f;
  float rmax[8], rsum[8];
#pragma unroll
  for (int g = 0; g < 8; ++g) { rmax[g] = -3.0e38f; rsum[g] = 0.0f; }

  const unsigned short* kbase = Kb + (size_t)(b * HK_ + hk) * T_ * D_;
  const unsigned short* vbase = Vt + (size_t)(b * HK_ + hk) * D_ * T_;

  for (int nt = 0; nt <= qt; ++nt) {
    const int kvBase = nt * 64;

    // prefetch next KV tile's K rows while we crunch this one
    if (nt < qt) {
      __builtin_prefetch(kbase + (size_t)(kvBase + 64 + lane) * D_, 0, 1);
      __builtin_prefetch(kbase + (size_t)(kvBase + 96 + lane) * D_, 0, 1);
    }

    // S = Q * K^T  (16 x 64 per wave)
    v8f s[4];
#pragma unroll
    for (int ns = 0; ns < 4; ++ns) s[ns] = (v8f)0.0f;
#pragma unroll
    for (int ns = 0; ns < 4; ++ns) {
      int kvcol = kvBase + ns * 16 + l15;
      const unsigned short* kp = kbase + (size_t)kvcol * D_ + lHi * 8;
#pragma unroll
      for (int ks = 0; ks < 4; ++ks) {
        Frag kf;
        kf.u[0] = *(const uint4*)(kp + ks * 32);
        kf.u[1] = *(const uint4*)(kp + ks * 32 + 16);
        s[ns] = wmma_bf16(qa[ks].v, kf.v, s[ns]);
      }
    }

    // causal mask on diagonal tile
    if (nt == qt) {
#pragma unroll
      for (int ns = 0; ns < 4; ++ns)
#pragma unroll
        for (int g = 0; g < 8; ++g) {
          int ng = kvBase + ns * 16 + l15;
          int mg = qRowBase + lHi * 8 + g;
          if (ng > mg) s[ns][g] = -1.0e30f;
        }
    }

    // online softmax per row (rows g of this lane's half-group)
#pragma unroll
    for (int g = 0; g < 8; ++g) {
      float m = s[0][g];
#pragma unroll
      for (int ns = 1; ns < 4; ++ns) m = fmaxf(m, s[ns][g]);
#pragma unroll
      for (int off = 1; off < 16; off <<= 1) m = fmaxf(m, __shfl_xor(m, off));
      float nm = fmaxf(rmax[g], m);
      float fc = __expf(rmax[g] - nm);
      float ts = 0.0f;
#pragma unroll
      for (int ns = 0; ns < 4; ++ns) {
        float pv = __expf(s[ns][g] - nm);
        s[ns][g] = pv;
        ts += pv;
      }
#pragma unroll
      for (int off = 1; off < 16; off <<= 1) ts += __shfl_xor(ts, off);
      rsum[g] = rsum[g] * fc + ts;
      rmax[g] = nm;
#pragma unroll
      for (int d8 = 0; d8 < 8; ++d8) o[d8][g] *= fc;
      // stage P (C-layout -> LDS row-major) for the P*V A-operand
#pragma unroll
      for (int ns = 0; ns < 4; ++ns)
        Pl[wave * 1024 + (lHi * 8 + g) * 64 + ns * 16 + l15] = f2bf(s[ns][g]);
    }
    __syncthreads();

    // P fragments (A-layout) from LDS
    Frag pa[2];
#pragma unroll
    for (int ks = 0; ks < 2; ++ks) {
      const unsigned short* pp = &Pl[wave * 1024 + l15 * 64 + ks * 32];
      pa[ks].u[0] = *(const uint4*)(pp + lHi * 8);
      pa[ks].u[1] = *(const uint4*)(pp + lHi * 8 + 16);
    }

    // O += P * V
#pragma unroll
    for (int d8 = 0; d8 < 8; ++d8) {
      int dcol = d8 * 16 + l15;
      const unsigned short* vp = vbase + (size_t)dcol * T_ + kvBase + lHi * 8;
#pragma unroll
      for (int ks = 0; ks < 2; ++ks) {
        Frag vfr;
        vfr.u[0] = *(const uint4*)(vp + ks * 32);
        vfr.u[1] = *(const uint4*)(vp + ks * 32 + 16);
        o[d8] = wmma_bf16(pa[ks].v, vfr.v, o[d8]);
      }
    }
    __syncthreads();
  }

  // normalize and store [B,T,H,D] bf16
  float inv[8];
#pragma unroll
  for (int g = 0; g < 8; ++g) inv[g] = 1.0f / rsum[g];
#pragma unroll
  for (int d8 = 0; d8 < 8; ++d8)
#pragma unroll
    for (int g = 0; g < 8; ++g) {
      int tq = qRowBase + lHi * 8 + g;
      Ob[((size_t)(b * T_ + tq) * H_ + h) * D_ + d8 * 16 + l15] =
          f2bf(o[d8][g] * inv[g]);
    }
}

// ---------------- host launch ----------------
extern "C" void kernel_launch(void* const* d_in, const int* in_sizes, int n_in,
                              void* d_out, int out_size, void* d_ws, size_t ws_size,
                              hipStream_t stream) {
  (void)in_sizes; (void)n_in; (void)out_size; (void)ws_size;
  const float* x    = (const float*)d_in[0];
  const float* wq   = (const float*)d_in[1];
  const float* wk   = (const float*)d_in[2];
  const float* wv   = (const float*)d_in[3];
  const float* wo   = (const float*)d_in[4];
  const float* cosT = (const float*)d_in[5];
  const float* sinT = (const float*)d_in[6];

  char* p = (char*)d_ws;
  auto alloc = [&](size_t bytes) -> char* {
    char* r = p;
    p += (bytes + 255) & ~(size_t)255;
    return r;
  };
  unsigned short* xb  = (unsigned short*)alloc((size_t)B_ * T_ * C_ * 2);
  unsigned short* wqb = (unsigned short*)alloc((size_t)QD_ * C_ * 2);
  unsigned short* wkb = (unsigned short*)alloc((size_t)KD_ * C_ * 2);
  unsigned short* wvb = (unsigned short*)alloc((size_t)KD_ * C_ * 2);
  unsigned short* wob = (unsigned short*)alloc((size_t)C_ * QD_ * 2);
  float*          qf  = (float*)alloc((size_t)B_ * T_ * QD_ * 4);
  float*          kf  = (float*)alloc((size_t)B_ * T_ * KD_ * 4);
  float*          vf  = (float*)alloc((size_t)B_ * T_ * KD_ * 4);
  unsigned short* qb  = (unsigned short*)alloc((size_t)B_ * T_ * QD_ * 2);
  unsigned short* kb  = (unsigned short*)alloc((size_t)B_ * HK_ * T_ * D_ * 2);
  unsigned short* vtb = (unsigned short*)alloc((size_t)B_ * HK_ * D_ * T_ * 2);
  unsigned short* ob  = (unsigned short*)alloc((size_t)B_ * T_ * QD_ * 2);

  auto cvt = [&](const float* s, unsigned short* d, long n) {
    long n4 = n / 4;
    cvt_f32_bf16<<<dim3((unsigned)((n4 + 255) / 256)), dim3(256), 0, stream>>>(s, d, n4);
  };
  cvt(x,  xb,  (long)B_ * T_ * C_);
  cvt(wq, wqb, (long)QD_ * C_);
  cvt(wk, wkb, (long)KD_ * C_);
  cvt(wv, wvb, (long)KD_ * C_);
  cvt(wo, wob, (long)C_ * QD_);

  const int M = B_ * T_;  // 2048
  gemm_bf16_tn<<<dim3(QD_ / 128, M / 128), 256, 0, stream>>>(xb, wqb, qf, M, QD_, C_);
  gemm_bf16_tn<<<dim3(KD_ / 128, M / 128), 256, 0, stream>>>(xb, wkb, kf, M, KD_, C_);
  gemm_bf16_tn<<<dim3(KD_ / 128, M / 128), 256, 0, stream>>>(xb, wvb, vf, M, KD_, C_);

  {
    long n = (long)B_ * T_ * H_ * (D_ / 2);
    rope_q_kernel<<<dim3((unsigned)((n + 255) / 256)), dim3(256), 0, stream>>>(qf, cosT, sinT, qb);
  }
  {
    long n = (long)B_ * T_ * HK_ * (D_ / 2);
    rope_k_kernel<<<dim3((unsigned)((n + 255) / 256)), dim3(256), 0, stream>>>(kf, cosT, sinT, kb);
  }
  {
    long n = (long)B_ * T_ * HK_ * D_;
    vt_pack_kernel<<<dim3((unsigned)((n + 255) / 256)), dim3(256), 0, stream>>>(vf, vtb);
  }

  flash_attn<<<dim3(T_ / 64, B_ * H_), dim3(128), 0, stream>>>(qb, kb, vtb, ob);

  gemm_bf16_tn<<<dim3(C_ / 128, M / 128), 256, 0, stream>>>(ob, wob, (float*)d_out,
                                                            M, C_, QD_);
}
